// TrajectoryEncoder_70540542870062
// MI455X (gfx1250) — matrix operand order repaired and checked by reading, not compiled, not run on hardware
//
#include <hip/hip_runtime.h>
#include <hip/hip_bf16.h>

// ---------------------------------------------------------------------------
// CDNA5 / gfx1250 implementation. wave32. All dense projections use
// v_wmma_f32_16x16x32_f16 (f16 inputs, f32 accumulate). Attention (spatial
// and map cross-attention) uses WMMA for both Q@K^T and P@V with an online
// softmax per 16-row block. GEMM staging uses GLOBAL_LOAD_ASYNC_TO_LDS with
// double-buffered LDS tiles and s_wait_asynccnt pipelining.
// ---------------------------------------------------------------------------

typedef __attribute__((ext_vector_type(16))) _Float16 v16h;
typedef __attribute__((ext_vector_type(8)))  _Float16 v8h;
typedef __attribute__((ext_vector_type(8)))  float    v8f;
typedef __attribute__((ext_vector_type(4)))  int      v4i;

union H16 { v16h v; v8h h[2]; };

// Problem constants (from reference)
#define Dm   256
#define Ed   512
#define Sd   16
#define Hh   8
#define DH   32
#define Bb   4
#define Tt   128
#define Nn   64
#define Ff   10
#define MTOK 256   // map tokens
#define MD   64
#define TOK  32768 // B*T*N

// ---- CDNA5 async memory -> LDS path (ASYNCcnt) -----------------------------
#if __has_builtin(__builtin_amdgcn_global_load_async_to_lds_b128) && \
    __has_builtin(__builtin_amdgcn_s_wait_asynccnt)
#define HAVE_GFX1250_ASYNC 1
#endif

#define AS1 __attribute__((address_space(1)))
#define AS3 __attribute__((address_space(3)))

__device__ __forceinline__ void async_cp_b128(const _Float16* g, _Float16* l) {
#ifdef HAVE_GFX1250_ASYNC
  // generic->AS3 via integer round trip: low 32 bits of a generic LDS address
  // are the LDS byte offset (aperture layout). Builtin params are v4i*.
  __builtin_amdgcn_global_load_async_to_lds_b128(
      (AS1 v4i*)(size_t)g, (AS3 v4i*)(unsigned)(size_t)l, 0, 0);
#else
  *(v8h*)l = *(const v8h*)g;
#endif
}

#ifdef HAVE_GFX1250_ASYNC
#define WAIT_ASYNC(n) __builtin_amdgcn_s_wait_asynccnt(n)
#else
#define WAIT_ASYNC(n) ((void)0)
#endif

__device__ __forceinline__ v8f wmma_f16(v16h a, v16h b, v8f c) {
  // 8 args: (neg_a, A, neg_b, B, c_mod, C, reuse_a, reuse_b)
  return __builtin_amdgcn_wmma_f32_16x16x32_f16(false, a, false, b, (short)0, c,
                                                false, false);
}

// A-matrix 16x32 f16 fragment (ISA 7.12.2): lane&15 = row M;
// halves 0..7 -> K = sel*8 + i, halves 8..15 -> K = 16 + sel*8 + i.
__device__ __forceinline__ v16h afrag(const _Float16* p0, int ld, int k0, int lane) {
  const _Float16* p = p0 + (size_t)(lane & 15) * ld + k0;
  int sel = lane >> 4;
  H16 u;
  u.h[0] = *(const v8h*)(p + sel * 8);
  u.h[1] = *(const v8h*)(p + 16 + sel * 8);
  return u.v;
}

// B-matrix 32x16 f16 fragment, source stored N-major (Bt[n][k]):
// lane&15 = column N; halves 0..7 -> K = sel*8 + i, halves 8..15 -> K = 16 + sel*8 + i.
__device__ __forceinline__ v16h bfrag(const _Float16* p0, int ld, int k0, int lane) {
  const _Float16* p = p0 + (size_t)(lane & 15) * ld + k0;
  int sel = lane >> 4;
  H16 u;
  u.h[0] = *(const v8h*)(p + sel * 8);
  u.h[1] = *(const v8h*)(p + 16 + sel * 8);
  return u.v;
}

__device__ __forceinline__ float silu_f(float v)  { return v / (1.f + __expf(-v)); }
__device__ __forceinline__ float softplus_f(float v) {
  return (v > 20.f) ? v : log1pf(__expf(v));
}

// ---------------------------------------------------------------------------
// Weight convert+transpose: W[K][N] f32 -> Wt[N][K] f16
// ---------------------------------------------------------------------------
__global__ __launch_bounds__(256) void cvt_wT(const float* __restrict__ W,
                                              _Float16* __restrict__ Wt,
                                              int Kd, int Nd) {
  int idx = blockIdx.x * 256 + threadIdx.x;
  if (idx < Kd * Nd) {
    int k = idx / Nd, n = idx % Nd;
    Wt[(size_t)n * Kd + k] = (_Float16)W[idx];
  }
}

// activations f32 -> f16 (no transpose)
__global__ __launch_bounds__(256) void cvt_h(const float* __restrict__ A,
                                             _Float16* __restrict__ Ah, int n) {
  int idx = blockIdx.x * 256 + threadIdx.x;
  if (idx < n) Ah[idx] = (_Float16)A[idx];
}

// ---------------------------------------------------------------------------
// Embed (K=10) + sinusoidal positional encoding over T.
// ---------------------------------------------------------------------------
__global__ __launch_bounds__(256) void embed_pe(const float* __restrict__ x,
                                                const float* __restrict__ W,
                                                const float* __restrict__ b,
                                                float* __restrict__ Hf,
                                                _Float16* __restrict__ Hhh) {
  int token = blockIdx.x;          // (b*T + t)*N + n
  int d = threadIdx.x;             // 0..255
  const float* xr = x + (size_t)token * Ff;
  float acc = b[d];
#pragma unroll
  for (int f = 0; f < Ff; ++f) acc += xr[f] * W[f * Dm + d];
  int t = (token / Nn) % Tt;
  float div = __expf((float)(d & ~1) * (-9.210340371976184f / (float)Dm)); // ln(1e4)
  float ang = (float)t * div;
  acc += (d & 1) ? __cosf(ang) : __sinf(ang);
  size_t o = (size_t)token * Dm + d;
  Hf[o] = acc;
  Hhh[o] = (_Float16)acc;
}

// ---------------------------------------------------------------------------
// Generic WMMA GEMM:  C[M][N] = act(A[M][K] @ Wt[N][K]^T + bias)
// Block tile 128x64, 8 waves (4x2), 32x32 per wave, K-step 32.
// Double-buffered LDS tiles filled by GLOBAL_LOAD_ASYNC_TO_LDS_B128;
// s_wait_asynccnt(3) keeps the next tile's 3 per-thread copies in flight
// while the current tile's WMMAs execute (async loads retire in order).
// Out-of-range A/B rows are clamped (their outputs are never stored), which
// keeps the per-wave async issue count uniform -> valid wait threshold.
// act: 0=none, 1=silu, 2=softplus. Cf / Chh each optional.
// ---------------------------------------------------------------------------
__global__ __launch_bounds__(256) void gemm_wmma(const _Float16* __restrict__ A,
                                                 const _Float16* __restrict__ Bt,
                                                 const float* __restrict__ bias,
                                                 float* __restrict__ Cf,
                                                 _Float16* __restrict__ Chh,
                                                 int Mdim, int Ndim, int Kdim,
                                                 int act) {
  __shared__ alignas(16) _Float16 As[2][128][40];
  __shared__ alignas(16) _Float16 Bs[2][64][40];
  int tid = threadIdx.x, lane = tid & 31, wave = tid >> 5;
  int bm = blockIdx.x * 128;
  int bn = blockIdx.y * 64;
  int wm = (wave >> 1) * 32, wn = (wave & 1) * 32;
  v8f acc[2][2] = {{{}, {}}, {{}, {}}};

  // Per-thread staging: two A chunks (rows sr, 64+sr) + one B chunk (row sr),
  // 16B (8 halves) each at column skk.
  int sr = tid >> 2;
  int skk = (tid & 3) * 8;
  int am0 = min(bm + sr, Mdim - 1);
  int am1 = min(bm + 64 + sr, Mdim - 1);
  int bnr = min(bn + sr, Ndim - 1);
  const _Float16* Ap0 = A + (size_t)am0 * Kdim + skk;
  const _Float16* Ap1 = A + (size_t)am1 * Kdim + skk;
  const _Float16* Bp  = Bt + (size_t)bnr * Kdim + skk;

  int nt = Kdim >> 5;  // K is a multiple of 32 for every projection here
  // stage tile 0 into buffer 0
  async_cp_b128(Ap0, &As[0][sr][skk]);
  async_cp_b128(Ap1, &As[0][64 + sr][skk]);
  async_cp_b128(Bp,  &Bs[0][sr][skk]);

  for (int t = 0; t < nt; ++t) {
    int k0 = t << 5;
    if (t + 1 < nt) {  // issue next tile, then wait for current (3 in flight)
      int nb = (t + 1) & 1;
      int nk = k0 + 32;
      async_cp_b128(Ap0 + nk, &As[nb][sr][skk]);
      async_cp_b128(Ap1 + nk, &As[nb][64 + sr][skk]);
      async_cp_b128(Bp + nk,  &Bs[nb][sr][skk]);
      WAIT_ASYNC(3);
    } else {
      WAIT_ASYNC(0);
    }
    if (t + 2 < nt)  // pull tile t+2 toward L2 (global_prefetch_b8)
      __builtin_prefetch(Ap0 + k0 + 64, 0, 1);
    __syncthreads();

    int cb = t & 1;
    v16h af[2], bf[2];
#pragma unroll
    for (int i = 0; i < 2; ++i) af[i] = afrag(&As[cb][wm + i * 16][0], 40, 0, lane);
#pragma unroll
    for (int j = 0; j < 2; ++j) bf[j] = bfrag(&Bs[cb][wn + j * 16][0], 40, 0, lane);
#pragma unroll
    for (int i = 0; i < 2; ++i)
#pragma unroll
      for (int j = 0; j < 2; ++j) acc[i][j] = wmma_f16(af[i], bf[j], acc[i][j]);
    __syncthreads();
  }

  int sel = lane >> 4;
  int ncol = lane & 15;
#pragma unroll
  for (int i = 0; i < 2; ++i)
#pragma unroll
    for (int j = 0; j < 2; ++j) {
      int gn = bn + wn + j * 16 + ncol;
      if (gn >= Ndim) continue;
      float bv = bias ? bias[gn] : 0.f;
#pragma unroll
      for (int r = 0; r < 8; ++r) {
        int gm = bm + wm + i * 16 + r + 8 * sel;
        if (gm >= Mdim) continue;
        float v = acc[i][j][r] + bv;
        if (act == 1) v = silu_f(v);
        else if (act == 2) v = softplus_f(v);
        size_t o = (size_t)gm * Ndim + gn;
        if (Cf) Cf[o] = v;
        if (Chh) Chh[o] = (_Float16)v;
      }
    }
}

// ---------------------------------------------------------------------------
// Causal depthwise conv (K=4) over T + SiLU.  uz: [TOK][2E], u = cols 0..E-1.
// One thread per (b,n,e), serial over t.
// ---------------------------------------------------------------------------
__global__ __launch_bounds__(256) void conv_silu(const float* __restrict__ uz,
                                                 const float* __restrict__ cW,
                                                 const float* __restrict__ cb,
                                                 float* __restrict__ Uf,
                                                 _Float16* __restrict__ Uh) {
  int idx = blockIdx.x * 256 + threadIdx.x;   // 0 .. B*N*E-1
  int e = idx % Ed;
  int bn = idx / Ed;
  int n = bn % Nn, b = bn / Nn;
  float w0 = cW[0 * Ed + e], w1 = cW[1 * Ed + e];
  float w2 = cW[2 * Ed + e], w3 = cW[3 * Ed + e];
  float bb = cb[e];
  float x0 = 0.f, x1 = 0.f, x2 = 0.f;
  for (int t = 0; t < Tt; ++t) {
    size_t tok = ((size_t)(b * Tt + t) * Nn + n);
    float x3 = uz[tok * (2 * Ed) + e];
    float a = bb + w0 * x0 + w1 * x1 + w2 * x2 + w3 * x3;  // causal: W[3]*x[t]
    a = silu_f(a);
    size_t o = tok * Ed + e;
    Uf[o] = a;
    Uh[o] = (_Float16)a;
    x0 = x1; x1 = x2; x2 = x3;
  }
}

// ---------------------------------------------------------------------------
// Mamba selective scan. Block per (b,n), 512 threads = e. S=16 state in regs.
// Writes y*silu(z) in f16 (feeds out_proj GEMM).
// ---------------------------------------------------------------------------
__global__ __launch_bounds__(512) void mamba_scan(const float* __restrict__ delta,
                                                  const float* __restrict__ Btm,
                                                  const float* __restrict__ Ctm,
                                                  const float* __restrict__ U,
                                                  const float* __restrict__ uz,
                                                  const float* __restrict__ A_log,
                                                  const float* __restrict__ Dp,
                                                  _Float16* __restrict__ Yh) {
  int b = blockIdx.x / Nn, n = blockIdx.x % Nn;
  int e = threadIdx.x;
  float A[Sd];
#pragma unroll
  for (int s = 0; s < Sd; ++s) A[s] = -__expf(A_log[e * Sd + s]);
  float Dv = Dp[e];
  float h[Sd];
#pragma unroll
  for (int s = 0; s < Sd; ++s) h[s] = 0.f;
  __shared__ float Bsm[Sd], Csm[Sd];
  for (int t = 0; t < Tt; ++t) {
    size_t tok = (size_t)(b * Tt + t) * Nn + n;
    if (threadIdx.x < Sd) Bsm[threadIdx.x] = Btm[tok * Sd + threadIdx.x];
    else if (threadIdx.x < 2 * Sd) Csm[threadIdx.x - Sd] = Ctm[tok * Sd + threadIdx.x - Sd];
    __syncthreads();
    float d = delta[tok * Ed + e];
    float u = U[tok * Ed + e];
    float du = d * u;
    float y = 0.f;
#pragma unroll
    for (int s = 0; s < Sd; ++s) {
      h[s] = __expf(d * A[s]) * h[s] + du * Bsm[s];
      y += h[s] * Csm[s];
    }
    y += u * Dv;
    float z = uz[tok * (2 * Ed) + Ed + e];
    y *= silu_f(z);
    Yh[tok * Ed + e] = (_Float16)y;
    __syncthreads();
  }
}

// ---------------------------------------------------------------------------
// Unified WMMA attention (flash-style online softmax).
// Block = (b,t) pair, wave = head. nkeys = 64 (spatial) or 256 (map).
// Q/K/V f16 [rows][256], head slice = cols h*32..h*32+31.
// Q and K fragments load straight from global (A/B layouts are contiguous
// 16B runs in row-major storage); V is transposed through LDS for the
// P@V B-fragment; P goes through LDS for the A-fragment relayout.
// ---------------------------------------------------------------------------
__global__ __launch_bounds__(256) void attn_wmma(const _Float16* __restrict__ Q,
                                                 const _Float16* __restrict__ Km,
                                                 const _Float16* __restrict__ Vm,
                                                 _Float16* __restrict__ O,
                                                 int nkeys) {
  __shared__ alignas(16) _Float16 VsT[8][32][72];  // per-wave V chunk, transposed
  __shared__ alignas(16) _Float16 Ps[8][16][72];   // per-wave P staging
  int bt = blockIdx.x;
  int wave = threadIdx.x >> 5, lane = threadIdx.x & 31;
  int sel = lane >> 4, ncol = lane & 15;
  int hc = wave * DH;
  size_t qbase = (size_t)bt * Nn;
  size_t kvbase = (nkeys == Nn) ? (size_t)bt * Nn : (size_t)(bt / Tt) * MTOK;
  const float scale = 0.17677669529663687f;  // 1/sqrt(32)
  int nchunks = nkeys / 64;

  for (int mb = 0; mb < 4; ++mb) {
    v8f o0 = {}, o1 = {};
    float rmax[8], rsum[8];
#pragma unroll
    for (int r = 0; r < 8; ++r) { rmax[r] = -1e30f; rsum[r] = 0.f; }

    for (int kc = 0; kc < nchunks; ++kc) {
      // stage V chunk transposed: VsT[k][n] = V[kvbase+kc*64+n][hc+k]
      for (int r = lane; r < 64; r += 32) {
        const _Float16* vp = Vm + (kvbase + kc * 64 + r) * Dm + hc;
#pragma unroll
        for (int c = 0; c < 32; c += 8) {
          v8h vv = *(const v8h*)(vp + c);
#pragma unroll
          for (int j = 0; j < 8; ++j) VsT[wave][c + j][r] = vv[j];
        }
      }
      __syncthreads();

      // scores: 16x64 = 4 WMMA (Q 16x32 @ K^T 32x16)
      v16h qa = afrag(Q + (qbase + mb * 16) * Dm + hc, Dm, 0, lane);
      v8f sacc[4];
#pragma unroll
      for (int nb = 0; nb < 4; ++nb) {
        v16h kb = bfrag(Km + (kvbase + kc * 64 + nb * 16) * Dm + hc, Dm, 0, lane);
        v8f z = {};
        sacc[nb] = wmma_f16(qa, kb, z);
      }

      // online softmax over this 64-key chunk (16-lane half-wave rows)
#pragma unroll
      for (int r = 0; r < 8; ++r) {
        float m = -1e30f;
#pragma unroll
        for (int nb = 0; nb < 4; ++nb) m = fmaxf(m, sacc[nb][r] * scale);
#pragma unroll
        for (int d = 1; d < 16; d <<= 1) m = fmaxf(m, __shfl_xor(m, d, 32));
        float nm = fmaxf(rmax[r], m);
        float alpha = __expf(rmax[r] - nm);
        rmax[r] = nm;
        float psum = 0.f;
#pragma unroll
        for (int nb = 0; nb < 4; ++nb) {
          float p = __expf(sacc[nb][r] * scale - nm);
          sacc[nb][r] = p;
          psum += p;
        }
#pragma unroll
        for (int d = 1; d < 16; d <<= 1) psum += __shfl_xor(psum, d, 32);
        rsum[r] = rsum[r] * alpha + psum;
        o0[r] *= alpha;
        o1[r] *= alpha;
      }

      // stage P (f16) for A-fragment relayout: element (r+8*sel, nb*16+ncol)
#pragma unroll
      for (int nb = 0; nb < 4; ++nb)
#pragma unroll
        for (int r = 0; r < 8; ++r)
          Ps[wave][r + 8 * sel][nb * 16 + ncol] = (_Float16)sacc[nb][r];
      __syncthreads();

      // O += P(16x64) @ V(64x32): 2 K-chunks x 2 N-tiles
#pragma unroll
      for (int k0 = 0; k0 < 64; k0 += 32) {
        v16h pa = afrag(&Ps[wave][0][0], 72, k0, lane);
        v16h vb0 = bfrag(&VsT[wave][0][0], 72, k0, lane);
        v16h vb1 = bfrag(&VsT[wave][16][0], 72, k0, lane);
        o0 = wmma_f16(pa, vb0, o0);
        o1 = wmma_f16(pa, vb1, o1);
      }
      __syncthreads();
    }

    // normalize and store head output (f16, feeds Wo GEMM)
#pragma unroll
    for (int r = 0; r < 8; ++r) {
      float inv = 1.f / rsum[r];
      size_t row = qbase + mb * 16 + r + 8 * sel;
      O[row * Dm + hc + ncol]      = (_Float16)(o0[r] * inv);
      O[row * Dm + hc + 16 + ncol] = (_Float16)(o1[r] * inv);
    }
  }
}

// ---------------------------------------------------------------------------
// Residual add + LayerNorm over D=256. One wave per row, 8 cols per lane.
// In-place safe on (Xf) since each thread reads its cols before writing.
// ---------------------------------------------------------------------------
__global__ __launch_bounds__(256) void add_ln(const float* __restrict__ Xf,
                                              const float* __restrict__ R,
                                              const float* __restrict__ g,
                                              const float* __restrict__ bta,
                                              float* __restrict__ Yf,
                                              _Float16* __restrict__ Yh) {
  int wave = threadIdx.x >> 5, lane = threadIdx.x & 31;
  size_t row = (size_t)blockIdx.x * 8 + wave;
  const float* xr = Xf + row * Dm;
  const float* rr = R + row * Dm;
  float v[8];
  float s = 0.f;
#pragma unroll
  for (int i = 0; i < 8; ++i) {
    v[i] = xr[lane + 32 * i] + rr[lane + 32 * i];
    s += v[i];
  }
#pragma unroll
  for (int d = 1; d < 32; d <<= 1) s += __shfl_xor(s, d, 32);
  float mu = s * (1.f / Dm);
  float var = 0.f;
#pragma unroll
  for (int i = 0; i < 8; ++i) { float t = v[i] - mu; var += t * t; }
#pragma unroll
  for (int d = 1; d < 32; d <<= 1) var += __shfl_xor(var, d, 32);
  float inv = rsqrtf(var * (1.f / Dm) + 1e-5f);
#pragma unroll
  for (int i = 0; i < 8; ++i) {
    int c = lane + 32 * i;
    float o = (v[i] - mu) * inv * g[c] + bta[c];
    Yf[row * Dm + c] = o;
    if (Yh) Yh[row * Dm + c] = (_Float16)o;
  }
}

// ---------------------------------------------------------------------------
// Host orchestration
// ---------------------------------------------------------------------------
static inline void launch_gemm(const _Float16* A, const _Float16* Bt, const float* bias,
                               float* Cf, _Float16* Ch, int M, int N, int K, int act,
                               hipStream_t s) {
  dim3 grid((M + 127) / 128, (N + 63) / 64);
  gemm_wmma<<<grid, 256, 0, s>>>(A, Bt, bias, Cf, Ch, M, N, K, act);
}
static inline void launch_cvtw(const float* W, _Float16* Wt, int K, int N, hipStream_t s) {
  cvt_wT<<<(K * N + 255) / 256, 256, 0, s>>>(W, Wt, K, N);
}

extern "C" void kernel_launch(void* const* d_in, const int* in_sizes, int n_in,
                              void* d_out, int out_size, void* d_ws, size_t ws_size,
                              hipStream_t stream) {
  (void)in_sizes; (void)n_in; (void)out_size; (void)ws_size;
#define PAR(i) ((const float*)d_in[(i)])
  const float* mapf     = PAR(1);
  const float* embed_W  = PAR(2);
  const float* embed_b  = PAR(3);
  const float* in_proj_W = PAR(4);
  const float* in_proj_b = PAR(5);
  const float* conv_W   = PAR(6);
  const float* conv_b   = PAR(7);
  const float* dt_W     = PAR(8);
  const float* dt_b     = PAR(9);
  const float* B_W      = PAR(10);
  const float* C_W      = PAR(11);
  const float* A_log    = PAR(12);
  const float* D_param  = PAR(13);
  const float* out_proj_W = PAR(14);
  const float* out_proj_b = PAR(15);
  const float* mamba_ln_g = PAR(16);
  const float* mamba_ln_b = PAR(17);
  // attn layers: 18..27 (layer0), 28..37 (layer1): Wq,Wk,Wv,Wo,bq,bk,bv,bo,ln_g,ln_b
  const float* map_W = PAR(38);
  const float* map_b = PAR(39);
  const float* mWq   = PAR(40);
  const float* mWk   = PAR(41);
  const float* mWv   = PAR(42);
  const float* mWo   = PAR(43);
  const float* m_ln_g = PAR(44);
  const float* m_ln_b = PAR(45);

  // --- workspace arena ---
  char* base = (char*)d_ws;
  size_t off = 0;
  auto alloc = [&](size_t bytes) -> void* {
    void* p = base + off;
    off += (bytes + 255) & ~(size_t)255;
    return p;
  };
  // f16 transposed weights
  _Float16* in_projT = (_Float16*)alloc((size_t)Dm * 2 * Ed * 2);
  _Float16* dtT      = (_Float16*)alloc((size_t)Ed * Ed * 2);
  _Float16* BWT      = (_Float16*)alloc((size_t)Ed * Sd * 2);
  _Float16* CWT      = (_Float16*)alloc((size_t)Ed * Sd * 2);
  _Float16* outT     = (_Float16*)alloc((size_t)Ed * Dm * 2);
  _Float16* attnT[2][4];
  for (int l = 0; l < 2; ++l)
    for (int w = 0; w < 4; ++w)
      attnT[l][w] = (_Float16*)alloc((size_t)Dm * Dm * 2);
  _Float16* mapWT = (_Float16*)alloc((size_t)MD * Dm * 2);
  _Float16* mWqT = (_Float16*)alloc((size_t)Dm * Dm * 2);
  _Float16* mWkT = (_Float16*)alloc((size_t)Dm * Dm * 2);
  _Float16* mWvT = (_Float16*)alloc((size_t)Dm * Dm * 2);
  _Float16* mWoT = (_Float16*)alloc((size_t)Dm * Dm * 2);
  // activations
  float*     hf  = (float*)alloc((size_t)TOK * Dm * 4);
  _Float16*  hhh = (_Float16*)alloc((size_t)TOK * Dm * 2);
  float*     uz  = (float*)alloc((size_t)TOK * 2 * Ed * 4);   // reused as q/k/v/o f16
  float*     uf  = (float*)alloc((size_t)TOK * Ed * 4);       // reused as tmp f32
  _Float16*  uh  = (_Float16*)alloc((size_t)TOK * Ed * 2);
  float*     dlt = (float*)alloc((size_t)TOK * Ed * 4);       // reused as map bufs
  float*     Btb = (float*)alloc((size_t)TOK * Sd * 4);
  float*     Ctb = (float*)alloc((size_t)TOK * Sd * 4);
  _Float16*  yh  = (_Float16*)alloc((size_t)TOK * Ed * 2);
  // aliases (live after the mamba scan retires uz/uf/dlt)
  _Float16* qh = (_Float16*)uz;
  _Float16* kh = qh + (size_t)TOK * Dm;
  _Float16* vh = kh + (size_t)TOK * Dm;
  _Float16* oh = vh + (size_t)TOK * Dm;
  float*    tmp = uf;
  _Float16* mfh = (_Float16*)dlt;
  _Float16* mh  = mfh + (size_t)Bb * MTOK * MD;
  _Float16* mkh = mh + (size_t)Bb * MTOK * Dm;
  _Float16* mvh = mkh + (size_t)Bb * MTOK * Dm;

  // --- weight conversion (f32 -> f16, transposed to N-major) ---
  launch_cvtw(in_proj_W, in_projT, Dm, 2 * Ed, stream);
  launch_cvtw(dt_W, dtT, Ed, Ed, stream);
  launch_cvtw(B_W, BWT, Ed, Sd, stream);
  launch_cvtw(C_W, CWT, Ed, Sd, stream);
  launch_cvtw(out_proj_W, outT, Ed, Dm, stream);
  for (int l = 0; l < 2; ++l)
    for (int w = 0; w < 4; ++w)
      launch_cvtw(PAR(18 + l * 10 + w), attnT[l][w], Dm, Dm, stream);
  launch_cvtw(map_W, mapWT, MD, Dm, stream);
  launch_cvtw(mWq, mWqT, Dm, Dm, stream);
  launch_cvtw(mWk, mWkT, Dm, Dm, stream);
  launch_cvtw(mWv, mWvT, Dm, Dm, stream);
  launch_cvtw(mWo, mWoT, Dm, Dm, stream);
  cvt_h<<<((Bb * MTOK * MD) + 255) / 256, 256, 0, stream>>>(mapf, mfh, Bb * MTOK * MD);

  // --- embed + positional encoding ---
  embed_pe<<<TOK, 256, 0, stream>>>(PAR(0), embed_W, embed_b, hf, hhh);

  // --- mamba block ---
  launch_gemm(hhh, in_projT, in_proj_b, uz, nullptr, TOK, 2 * Ed, Dm, 0, stream);
  conv_silu<<<(Bb * Nn * Ed) / 256, 256, 0, stream>>>(uz, conv_W, conv_b, uf, uh);
  launch_gemm(uh, dtT, dt_b, dlt, nullptr, TOK, Ed, Ed, 2, stream);       // softplus
  launch_gemm(uh, BWT, nullptr, Btb, nullptr, TOK, Sd, Ed, 0, stream);
  launch_gemm(uh, CWT, nullptr, Ctb, nullptr, TOK, Sd, Ed, 0, stream);
  mamba_scan<<<Bb * Nn, Ed, 0, stream>>>(dlt, Btb, Ctb, uf, uz, A_log, D_param, yh);
  launch_gemm(yh, outT, out_proj_b, tmp, nullptr, TOK, Dm, Ed, 0, stream);
  add_ln<<<TOK / 8, 256, 0, stream>>>(hf, tmp, mamba_ln_g, mamba_ln_b, hf, hhh);

  // --- spatial MHA layers ---
  for (int l = 0; l < 2; ++l) {
    const float* bq = PAR(18 + l * 10 + 4);
    const float* bk = PAR(18 + l * 10 + 5);
    const float* bv = PAR(18 + l * 10 + 6);
    const float* bo = PAR(18 + l * 10 + 7);
    const float* lg = PAR(18 + l * 10 + 8);
    const float* lb = PAR(18 + l * 10 + 9);
    launch_gemm(hhh, attnT[l][0], bq, nullptr, qh, TOK, Dm, Dm, 0, stream);
    launch_gemm(hhh, attnT[l][1], bk, nullptr, kh, TOK, Dm, Dm, 0, stream);
    launch_gemm(hhh, attnT[l][2], bv, nullptr, vh, TOK, Dm, Dm, 0, stream);
    attn_wmma<<<Bb * Tt, 256, 0, stream>>>(qh, kh, vh, oh, Nn);
    launch_gemm(oh, attnT[l][3], bo, tmp, nullptr, TOK, Dm, Dm, 0, stream);
    add_ln<<<TOK / 8, 256, 0, stream>>>(hf, tmp, lg, lb, hf, hhh);
  }

  // --- map cross-attention fuse ---
  launch_gemm(mfh, mapWT, map_b, nullptr, mh, Bb * MTOK, Dm, MD, 0, stream);
  launch_gemm(hhh, mWqT, nullptr, nullptr, qh, TOK, Dm, Dm, 0, stream);
  launch_gemm(mh, mWkT, nullptr, nullptr, mkh, Bb * MTOK, Dm, Dm, 0, stream);
  launch_gemm(mh, mWvT, nullptr, nullptr, mvh, Bb * MTOK, Dm, Dm, 0, stream);
  attn_wmma<<<Bb * Tt, 256, 0, stream>>>(qh, mkh, mvh, oh, MTOK);
  launch_gemm(oh, mWoT, nullptr, tmp, nullptr, TOK, Dm, Dm, 0, stream);
  add_ln<<<TOK / 8, 256, 0, stream>>>(hf, tmp, m_ln_g, m_ln_b, (float*)d_out, nullptr);
#undef PAR
}
